// RopeSelfAttention_79405355369016
// MI455X (gfx1250) — compile-verified, matrix operand order checked by reading, not run on hardware
//
#include <hip/hip_runtime.h>
#include <hip/hip_bf16.h>

// ---------------------------------------------------------------------------
// RoPE self-attention forward for MI455X (gfx1250, wave32, WMMA bf16).
//   B=4, S=2048, E=1024, H=16, hd=64
// Pipeline: cvt(fp32->bf16) x3 -> qkv_rope_gemm -> flash_attn -> proj_gemm
// All matmuls on v_wmma_f32_16x16x32_bf16 (fp32 accumulate); the GEMM hot
// loops read pre-converted bf16 so no v_cvt appears between WMMAs.
// ---------------------------------------------------------------------------

typedef __bf16 bf16;
typedef __attribute__((ext_vector_type(16))) __bf16 v16bf;
typedef __attribute__((ext_vector_type(8)))  __bf16 v8bf;
typedef __attribute__((ext_vector_type(8)))  float  v8f;

#define S_LEN 2048
#define EMB   1024
#define NHEAD 16
#define HDIM  64

__device__ __forceinline__ v8f wmma_bf16(v16bf a, v16bf b, v8f c) {
  // 8 args: (neg_a, A, neg_b, B, c_mod, C, reuse_a, reuse_b)
  return __builtin_amdgcn_wmma_f32_16x16x32_bf16(false, a, false, b,
                                                 (short)0, c, false, false);
}

__device__ __forceinline__ v16bf ld16bf(const bf16* __restrict__ p) {
  return *(const v16bf*)p;   // 32B -> two global_load_b128 halves
}

// ---------------------------------------------------------------------------
// Kernel 0: streaming fp32 -> bf16 conversion (8 elements / thread).
// ---------------------------------------------------------------------------
__global__ __launch_bounds__(256, 1)
void cvt_f32_bf16(const float* __restrict__ src, bf16* __restrict__ dst, int n) {
  int i = (blockIdx.x * 256 + threadIdx.x) * 8;
  if (i >= n) return;
  const float4* p = (const float4*)(src + i);
  float4 a = p[0], b = p[1];
  v8bf r;
  r[0] = (__bf16)a.x; r[1] = (__bf16)a.y; r[2] = (__bf16)a.z; r[3] = (__bf16)a.w;
  r[4] = (__bf16)b.x; r[5] = (__bf16)b.y; r[6] = (__bf16)b.z; r[7] = (__bf16)b.w;
  *(v8bf*)(dst + i) = r;
}

// ---------------------------------------------------------------------------
// Kernel 1: qkv = hidden @ qkv_w^T + qkv_b, fused RoPE on q/k in the epilogue.
// Block tile 128(M) x 128(N); wave tile 32 x 64 (= one full head wide).
// Outputs: q,k as bf16 [B,H,S,hd]; v as bf16 [B,H,hd,S] (transposed for PV).
// ---------------------------------------------------------------------------
__global__ __launch_bounds__(256, 1)
void qkv_rope_gemm(const bf16* __restrict__ A,     // [8192,1024] hidden (bf16)
                   const bf16* __restrict__ W,     // [3072,1024] qkv_w (bf16)
                   const float* __restrict__ bias, // [3072]
                   bf16* __restrict__ qo,          // [B,H,S,hd]
                   bf16* __restrict__ ko,          // [B,H,S,hd]
                   bf16* __restrict__ vt) {        // [B,H,hd,S]
  const int lane  = threadIdx.x & 31;
  const int wave  = threadIdx.x >> 5;
  const int mbase = blockIdx.x * 128 + (wave & 3) * 32;
  const int nbase = blockIdx.y * 128 + (wave >> 2) * 64;
  const int col   = lane & 15;
  const int koff  = (lane >> 4) * 16;   // which half of the K=32 slice this lane carries

  v8f acc[2][4] = {};
  const bf16* arow0 = A + (size_t)(mbase + col) * EMB;
  const bf16* arow1 = arow0 + 16 * EMB;

  for (int kk = 0; kk < EMB; kk += 32) {
    __builtin_prefetch(arow0 + kk + 256, 0, 1);
    v16bf af0 = ld16bf(arow0 + kk + koff);
    v16bf af1 = ld16bf(arow1 + kk + koff);
#pragma unroll
    for (int t = 0; t < 4; ++t) {
      v16bf wf = ld16bf(W + (size_t)(nbase + t * 16 + col) * EMB + kk + koff);
      acc[0][t] = wmma_bf16(af0, wf, acc[0][t]);
      acc[1][t] = wmma_bf16(af1, wf, acc[1][t]);
    }
  }

  // ---- epilogue: bias, RoPE (q/k), scatter to attention-friendly layouts ----
  const int which = nbase >> 10;               // 0=q 1=k 2=v
  const int h     = (nbase & 1023) >> 6;       // wave covers one full head (d 0..63)

#pragma unroll
  for (int mi = 0; mi < 2; ++mi) {
    const int rbase = mbase + mi * 16 + ((lane >> 4) << 3);
#pragma unroll
    for (int t = 0; t < 4; ++t) {
      float bv = bias[nbase + t * 16 + col];
#pragma unroll
      for (int r = 0; r < 8; ++r) acc[mi][t][r] += bv;
    }

    if (which < 2) {
      // RoPE: pair (d, d+32) == accumulator sub-tiles (t, t+2)
      bf16* outp = (which == 0) ? qo : ko;
#pragma unroll
      for (int t = 0; t < 2; ++t) {
        const int d1 = t * 16 + col;                          // 0..31
        const float inv_freq = __powf(10000.0f, -(float)d1 * (1.0f / 32.0f));
#pragma unroll
        for (int r = 0; r < 8; ++r) {
          const int s = (rbase + r) & (S_LEN - 1);
          float sn, cs;
          __sincosf((float)s * inv_freq, &sn, &cs);
          float x1 = acc[mi][t][r], x2 = acc[mi][t + 2][r];
          acc[mi][t][r]     = x1 * cs - x2 * sn;
          acc[mi][t + 2][r] = x2 * cs + x1 * sn;
        }
      }
#pragma unroll
      for (int t = 0; t < 4; ++t) {
        const int d = t * 16 + col;
#pragma unroll
        for (int r = 0; r < 8; ++r) {
          const int m = rbase + r;
          const int b = m >> 11;
          const int s = m & (S_LEN - 1);
          outp[(((size_t)(b * NHEAD + h)) * S_LEN + s) * HDIM + d] =
              (bf16)acc[mi][t][r];
        }
      }
    } else {
      // V stored transposed: vt[b,h,d,s]
#pragma unroll
      for (int t = 0; t < 4; ++t) {
        const int d = t * 16 + col;
#pragma unroll
        for (int r = 0; r < 8; ++r) {
          const int m = rbase + r;
          const int b = m >> 11;
          const int s = m & (S_LEN - 1);
          vt[(((size_t)(b * NHEAD + h)) * HDIM + d) * S_LEN + s] =
              (bf16)acc[mi][t][r];
        }
      }
    }
  }
}

// ---------------------------------------------------------------------------
// Kernel 2: flash attention. grid = (S/128, B*H); each wave owns 16 query
// rows x full head dim. Key tiles of 32; online softmax in fp32; P tile is
// re-layouted (C-layout -> A-layout) through a private LDS scratch per wave.
// ---------------------------------------------------------------------------
__global__ __launch_bounds__(256, 1)
void flash_attn(const bf16* __restrict__ q,   // [B,H,S,hd]
                const bf16* __restrict__ k,   // [B,H,S,hd]
                const bf16* __restrict__ vt,  // [B,H,hd,S]
                bf16* __restrict__ ao) {      // [B,S,E]
  __shared__ __align__(64) bf16 pbuf[8][16 * 32];

  const int lane = threadIdx.x & 31;
  const int wave = threadIdx.x >> 5;
  const int bh   = blockIdx.y;
  const int b    = bh >> 4, h = bh & 15;
  const int col  = lane & 15;
  const int dch  = (lane >> 4) * 16;
  const int qrow = blockIdx.x * 128 + wave * 16 + col;

  const bf16* qb = q  + (size_t)bh * S_LEN * HDIM;
  const bf16* kb = k  + (size_t)bh * S_LEN * HDIM;
  const bf16* vb = vt + (size_t)bh * HDIM * S_LEN;

  // Q fragments, resident in registers for the whole kernel.
  const v16bf a0 = ld16bf(qb + (size_t)qrow * HDIM + dch);
  const v16bf a1 = ld16bf(qb + (size_t)qrow * HDIM + dch + 32);

  float mstat[8], lstat[8];
  v8f o[4] = {};
#pragma unroll
  for (int r = 0; r < 8; ++r) { mstat[r] = -3.0e38f; lstat[r] = 0.0f; }

  bf16* myp = &pbuf[wave][0];

  for (int kt = 0; kt < S_LEN; kt += 32) {
    // ---- scores for keys [kt, kt+32): two 16-key tiles, contract over d ----
    v8f s0 = {}, s1 = {};
    const bf16* k0 = kb + (size_t)(kt + col) * HDIM + dch;
    s0 = wmma_bf16(a0, ld16bf(k0), s0);
    s0 = wmma_bf16(a1, ld16bf(k0 + 32), s0);
    const bf16* k1 = k0 + 16 * HDIM;
    s1 = wmma_bf16(a0, ld16bf(k1), s1);
    s1 = wmma_bf16(a1, ld16bf(k1 + 32), s1);

    // ---- online softmax (fp32). Score C-layout rows == O C-layout rows. ----
#pragma unroll
    for (int r = 0; r < 8; ++r) {
      float x0 = s0[r] * 0.125f;   // 1/sqrt(64)
      float x1 = s1[r] * 0.125f;
      float rm = fmaxf(x0, x1);
      rm = fmaxf(rm, __shfl_xor(rm, 1));
      rm = fmaxf(rm, __shfl_xor(rm, 2));
      rm = fmaxf(rm, __shfl_xor(rm, 4));
      rm = fmaxf(rm, __shfl_xor(rm, 8));
      float mnew = fmaxf(mstat[r], rm);
      float sc   = __expf(mstat[r] - mnew);
      float p0   = __expf(x0 - mnew);
      float p1   = __expf(x1 - mnew);
      float rs = p0 + p1;
      rs += __shfl_xor(rs, 1);
      rs += __shfl_xor(rs, 2);
      rs += __shfl_xor(rs, 4);
      rs += __shfl_xor(rs, 8);
      lstat[r] = lstat[r] * sc + rs;
      mstat[r] = mnew;
      o[0][r] *= sc; o[1][r] *= sc; o[2][r] *= sc; o[3][r] *= sc;
      // pack P into per-wave LDS tile [16 rows][32 keys] (C -> row-major)
      const int prow = r + ((lane >> 4) << 3);
      myp[prow * 32 + col]      = (bf16)p0;
      myp[prow * 32 + col + 16] = (bf16)p1;
    }

    // ---- O += P @ V : A = P (16x32 from LDS), B = V^T rows (contiguous) ----
    v16bf pa = *(const v16bf*)&myp[(lane & 15) * 32 + dch];
#pragma unroll
    for (int dt = 0; dt < 4; ++dt) {
      const bf16* vp = vb + (size_t)(dt * 16 + col) * S_LEN + kt + dch;
      o[dt] = wmma_bf16(pa, ld16bf(vp), o[dt]);
    }
  }

  // ---- normalize and store to [B,S,E] (bf16 input for the proj GEMM) ----
#pragma unroll
  for (int r = 0; r < 8; ++r) {
    float inv = 1.0f / lstat[r];
    o[0][r] *= inv; o[1][r] *= inv; o[2][r] *= inv; o[3][r] *= inv;
  }
  const int rbase = blockIdx.x * 128 + wave * 16 + ((lane >> 4) << 3);
#pragma unroll
  for (int dt = 0; dt < 4; ++dt) {
    const int e = h * HDIM + dt * 16 + col;
#pragma unroll
    for (int r = 0; r < 8; ++r) {
      const int s = rbase + r;
      ao[((size_t)(b * S_LEN + s)) * EMB + e] = (bf16)o[dt][r];
    }
  }
}

// ---------------------------------------------------------------------------
// Kernel 3: out = attn @ proj_w^T + proj_b  (bf16 x bf16 -> fp32)
// Block tile 128 x 128; wave tile 32 x 64.
// ---------------------------------------------------------------------------
__global__ __launch_bounds__(256, 1)
void proj_gemm(const bf16* __restrict__ A,     // [8192,1024]
               const bf16* __restrict__ W,     // [1024,1024] (bf16)
               const float* __restrict__ bias, // [1024]
               float* __restrict__ out) {      // [8192,1024]
  const int lane  = threadIdx.x & 31;
  const int wave  = threadIdx.x >> 5;
  const int mbase = blockIdx.x * 128 + (wave & 3) * 32;
  const int nbase = blockIdx.y * 128 + (wave >> 2) * 64;
  const int col   = lane & 15;
  const int koff  = (lane >> 4) * 16;

  v8f acc[2][4] = {};
  const bf16* arow0 = A + (size_t)(mbase + col) * EMB;
  const bf16* arow1 = arow0 + 16 * EMB;

  for (int kk = 0; kk < EMB; kk += 32) {
    __builtin_prefetch(arow0 + kk + 256, 0, 1);
    v16bf af0 = ld16bf(arow0 + kk + koff);
    v16bf af1 = ld16bf(arow1 + kk + koff);
#pragma unroll
    for (int t = 0; t < 4; ++t) {
      v16bf wf = ld16bf(W + (size_t)(nbase + t * 16 + col) * EMB + kk + koff);
      acc[0][t] = wmma_bf16(af0, wf, acc[0][t]);
      acc[1][t] = wmma_bf16(af1, wf, acc[1][t]);
    }
  }

#pragma unroll
  for (int mi = 0; mi < 2; ++mi) {
    const int rbase = mbase + mi * 16 + ((lane >> 4) << 3);
#pragma unroll
    for (int t = 0; t < 4; ++t) {
      const int n = nbase + t * 16 + col;
      const float bv = bias[n];
#pragma unroll
      for (int r = 0; r < 8; ++r)
        out[(size_t)(rbase + r) * EMB + n] = acc[mi][t][r] + bv;
    }
  }
}

// ---------------------------------------------------------------------------
extern "C" void kernel_launch(void* const* d_in, const int* in_sizes, int n_in,
                              void* d_out, int out_size, void* d_ws, size_t ws_size,
                              hipStream_t stream) {
  (void)in_sizes; (void)n_in; (void)out_size; (void)ws_size;
  const float* hidden = (const float*)d_in[0];
  const float* qkv_w  = (const float*)d_in[1];
  const float* qkv_b  = (const float*)d_in[2];
  const float* proj_w = (const float*)d_in[3];
  const float* proj_b = (const float*)d_in[4];
  float* out = (float*)d_out;

  // Workspace layout (bf16 elements):
  //   hb  [8192,1024]  hidden
  //   wq  [3072,1024]  qkv_w
  //   wp  [1024,1024]  proj_w
  //   q,k [B,H,S,hd]   vt [B,H,hd,S]   ao [B,S,E]
  const size_t N_H  = (size_t)8192 * 1024;   // 8.39M
  const size_t N_WQ = (size_t)3072 * 1024;   // 3.15M
  const size_t N_WP = (size_t)1024 * 1024;   // 1.05M
  bf16* hb = (bf16*)d_ws;
  bf16* wq = hb + N_H;
  bf16* wp = wq + N_WQ;
  bf16* q  = wp + N_WP;
  bf16* k  = q  + N_H;
  bf16* vt = k  + N_H;
  bf16* ao = vt + N_H;            // total ~92 MB of workspace

  const int CT = 256 * 8;
  cvt_f32_bf16<<<(int)(N_H  / CT), 256, 0, stream>>>(hidden, hb, (int)N_H);
  cvt_f32_bf16<<<(int)(N_WQ / CT), 256, 0, stream>>>(qkv_w,  wq, (int)N_WQ);
  cvt_f32_bf16<<<(int)(N_WP / CT), 256, 0, stream>>>(proj_w, wp, (int)N_WP);

  qkv_rope_gemm<<<dim3(64, 24), 256, 0, stream>>>(hb, wq, qkv_b, q, k, vt);
  flash_attn  <<<dim3(16, 64), 256, 0, stream>>>(q, k, vt, ao);
  proj_gemm   <<<dim3(64, 8),  256, 0, stream>>>(ao, wp, proj_b, out);
}
